// mamba_encoder_73289321939609
// MI455X (gfx1250) — compile-verified, matrix operand order checked by reading, not run on hardware
//
#include <hip/hip_runtime.h>
#include <hip/hip_bf16.h>
#include <math.h>

// ---------------------------------------------------------------------------
// Mamba block + MLP for MI455X (gfx1250, wave32, WMMA).
// Big GEMMs: LDS-tiled, double-buffered, staged with CDNA5 async
// global->LDS DMA (global_load_async_to_lds_b128, ASYNCcnt).
// All GEMMs: f16 inputs, f32 accumulation via v_wmma_f32_16x16x32_f16.
// ---------------------------------------------------------------------------

typedef _Float16 half_t;
typedef __attribute__((ext_vector_type(16))) _Float16 v16h;
typedef __attribute__((ext_vector_type(8)))  float    v8f;

#define D_MODEL 1024
#define D_INNER 2048
#define NSTATE  16
#define DTRANK  64
#define NB      4
#define SEQL    2048
#define NTOK    (NB * SEQL)   // 8192 tokens

#define ASYNC_STAGE 1         // CDNA5 async global->LDS DMA staging

// ---------------------------- f32 -> f16 convert ---------------------------
__global__ void k_f32_to_f16(const float* __restrict__ in, half_t* __restrict__ out, int n) {
  int i = blockIdx.x * blockDim.x + threadIdx.x;
  if (i < n) out[i] = (half_t)in[i];
}

// ------------------------------- RMSNorm -----------------------------------
__global__ void k_rmsnorm(const float* __restrict__ x, const float* __restrict__ w,
                          half_t* __restrict__ out, int D) {
  const int t = blockIdx.x;
  const float* row = x + (size_t)t * D;
  float ss = 0.f;
  for (int i = threadIdx.x; i < D; i += blockDim.x) { float v = row[i]; ss += v * v; }
  #pragma unroll
  for (int off = 16; off > 0; off >>= 1) ss += __shfl_xor(ss, off, 32);
  __shared__ float red[8];
  if ((threadIdx.x & 31) == 0) red[threadIdx.x >> 5] = ss;
  __syncthreads();
  float tot = 0.f;
  #pragma unroll
  for (int i = 0; i < 8; ++i) tot += red[i];
  const float inv = rsqrtf(tot / (float)D + 1e-6f);
  half_t* orow = out + (size_t)t * D;
  for (int i = threadIdx.x; i < D; i += blockDim.x)
    orow[i] = (half_t)(w[i] * row[i] * inv);
}

// ----------------------------- fragment loads ------------------------------
// Wave32 16-bit fragment layout: lane r holds K {0..7,16..23} (lanes 0-15)
// or K {8..15,24..31} (lanes 16-31) -> two contiguous 16-byte loads.
__device__ __forceinline__ v16h load_frag(const half_t* __restrict__ p) {
  union { v16h v; uint4 q[2]; } f;
  f.q[0] = *(const uint4*)(p);
  f.q[1] = *(const uint4*)(p + 16);
  return f.v;
}

// ----------------------- async global->LDS staging -------------------------
__device__ __forceinline__ void stage_b128(const half_t* __restrict__ g, half_t* l) {
#if ASYNC_STAGE
  // VDST = per-lane LDS byte address (low 32 bits of generic LDS address),
  // VADDR = 64-bit global address, saddr = off. Tracked with ASYNCcnt.
  unsigned loff = (unsigned)(size_t)l;
  asm volatile("global_load_async_to_lds_b128 %0, %1, off"
               :: "v"(loff), "v"(g) : "memory");
#else
  *(uint4*)l = *(const uint4*)g;
#endif
}

__device__ __forceinline__ void stage_fence() {
#if ASYNC_STAGE
  asm volatile("s_wait_asynccnt 0" ::: "memory");
#endif
  __syncthreads();
}

// ---------------- LDS-tiled WMMA GEMM (Y = A * W^T), big shapes ------------
// A: [M,K] f16 row-major, W: [N,K] f16 row-major. Requires M,N % 128 == 0,
// K % 64 == 0. Block: 256 threads = 8 waves (2M x 4N), tile 128x128, BK=64,
// double-buffered LDS staged via async DMA.
enum { EPI_F32 = 0, EPI_DBC, EPI_SOFTPLUS, EPI_RES, EPI_RELU, EPI_BIASRES };

#define BK 64
#define LDS_STRIDE 72   // padded row stride (halves): conflict-free, 16B aligned

template <int EPI>
__global__ __launch_bounds__(256)
void k_gemm_lds(const half_t* __restrict__ A, const half_t* __restrict__ W,
                float* __restrict__ outF, half_t* __restrict__ outH,
                const float* __restrict__ bias, const float* __restrict__ res,
                int M, int N, int K) {
  __shared__ __align__(16) half_t shA[2][128 * LDS_STRIDE];
  __shared__ __align__(16) half_t shB[2][128 * LDS_STRIDE];

  const int nblocks = N >> 7;
  const int mblk = blockIdx.x / nblocks;
  const int nblk = blockIdx.x - mblk * nblocks;
  const int m0 = mblk << 7, n0 = nblk << 7;
  const int tid = threadIdx.x;

  const half_t* Ag = A + (size_t)m0 * K;
  const half_t* Wg = W + (size_t)n0 * K;

  // stage one 128x64 A tile + 128x64 B tile: 1024 b128 transfers each way,
  // 4 A + 4 B per thread, row-contiguous for coalescing.
  auto stage = [&](int buf, int k0) {
    #pragma unroll
    for (int t = 0; t < 4; ++t) {
      const int id  = tid + t * 256;   // 0..1023
      const int row = id >> 3;
      const int ch  = (id & 7) * 8;
      stage_b128(Ag + (size_t)row * K + k0 + ch, &shA[buf][row * LDS_STRIDE + ch]);
      stage_b128(Wg + (size_t)row * K + k0 + ch, &shB[buf][row * LDS_STRIDE + ch]);
    }
  };

  const int wave = tid >> 5;
  const int lane = tid & 31;
  const int hf = lane >> 4;
  const int r  = lane & 15;
  const int wm = (wave & 1) * 64;   // wave M offset within block tile
  const int wn = (wave >> 1) * 32;  // wave N offset within block tile

  v8f acc[4][2] = {};

  stage(0, 0);
  stage_fence();

  const int ksteps = K >> 6;
  for (int kt = 0; kt < ksteps; ++kt) {
    if (kt + 1 < ksteps) stage((kt & 1) ^ 1, (kt + 1) << 6);  // prefetch next tile
    const half_t* a_base = &shA[kt & 1][0];
    const half_t* b_base = &shB[kt & 1][0];
    #pragma unroll
    for (int ks = 0; ks < BK; ks += 32) {
      v16h bf0 = load_frag(b_base + (wn + r) * LDS_STRIDE + ks + hf * 8);
      v16h bf1 = load_frag(b_base + (wn + 16 + r) * LDS_STRIDE + ks + hf * 8);
      #pragma unroll
      for (int sub = 0; sub < 4; ++sub) {
        v16h af = load_frag(a_base + (wm + sub * 16 + r) * LDS_STRIDE + ks + hf * 8);
        acc[sub][0] = __builtin_amdgcn_wmma_f32_16x16x32_f16(false, af, false, bf0, (short)0, acc[sub][0], false, false);
        acc[sub][1] = __builtin_amdgcn_wmma_f32_16x16x32_f16(false, af, false, bf1, (short)0, acc[sub][1], false, false);
      }
    }
    stage_fence();  // async tile landed + all waves done reading old buffer
  }

  #pragma unroll
  for (int nn = 0; nn < 2; ++nn) {
    const int col = n0 + wn + nn * 16 + r;
    #pragma unroll
    for (int sub = 0; sub < 4; ++sub) {
      #pragma unroll
      for (int j = 0; j < 8; ++j) {
        const int row = m0 + wm + sub * 16 + hf * 8 + j;   // VGPR j -> M = j + 8*hf
        float v = acc[sub][nn][j];
        const size_t o = (size_t)row * N + col;
        if (EPI == EPI_F32) {
          outF[o] = v;
        } else if (EPI == EPI_RES) {
          outF[o] = v + res[o];
        } else if (EPI == EPI_RELU) {
          float b = v + bias[col];
          outH[o] = (half_t)(b > 0.f ? b : 0.f);
        } else { // EPI_BIASRES
          outF[o] = v + bias[col] + res[o];
        }
      }
    }
  }
}

// ----------------- direct WMMA GEMM for small/odd shapes -------------------
// wave computes 64x16 strip straight from global (x_proj N=96, dt_proj K=64)
template <int EPI>
__global__ void k_gemm_wmma(const half_t* __restrict__ A, const half_t* __restrict__ W,
                            float* __restrict__ outF, half_t* __restrict__ outH,
                            const float* __restrict__ bias, const float* __restrict__ res,
                            int M, int N, int K) {
  const int wave   = (blockIdx.x * blockDim.x + threadIdx.x) >> 5;
  const int lane   = threadIdx.x & 31;
  const int ntiles = N >> 4;
  const int mtile  = wave / ntiles;
  const int ntile  = wave - mtile * ntiles;
  if (mtile >= (M >> 6)) return;
  const int m0 = mtile << 6;
  const int n0 = ntile << 4;
  const int hf = lane >> 4;
  const int r  = lane & 15;

  const half_t* wp = W + (size_t)(n0 + r) * K + hf * 8;
  const half_t* ap = A + (size_t)(m0 + r) * K + hf * 8;
  const size_t a16 = (size_t)16 * K;

  v8f acc0 = {}, acc1 = {}, acc2 = {}, acc3 = {};
  for (int k = 0; k < K; k += 32) {
    v16h bf = load_frag(wp + k);
    v16h a0 = load_frag(ap + k);
    v16h a1 = load_frag(ap + a16 + k);
    v16h a2 = load_frag(ap + 2 * a16 + k);
    v16h a3 = load_frag(ap + 3 * a16 + k);
    acc0 = __builtin_amdgcn_wmma_f32_16x16x32_f16(false, a0, false, bf, (short)0, acc0, false, false);
    acc1 = __builtin_amdgcn_wmma_f32_16x16x32_f16(false, a1, false, bf, (short)0, acc1, false, false);
    acc2 = __builtin_amdgcn_wmma_f32_16x16x32_f16(false, a2, false, bf, (short)0, acc2, false, false);
    acc3 = __builtin_amdgcn_wmma_f32_16x16x32_f16(false, a3, false, bf, (short)0, acc3, false, false);
  }

  v8f accs[4] = {acc0, acc1, acc2, acc3};
  const int col = n0 + r;
  #pragma unroll
  for (int sub = 0; sub < 4; ++sub) {
    #pragma unroll
    for (int j = 0; j < 8; ++j) {
      const int row = m0 + sub * 16 + hf * 8 + j;
      float v = accs[sub][j];
      const size_t o = (size_t)row * N + col;
      if (EPI == EPI_F32) {
        outF[o] = v;
      } else if (EPI == EPI_DBC) {
        outF[o] = v;                                                     // dbc f32 [row,96]
        if (col < DTRANK) outH[(size_t)row * DTRANK + col] = (half_t)v;  // dt part f16
      } else if (EPI == EPI_SOFTPLUS) {
        float b = v + bias[col];
        outF[o] = (b > 20.f) ? b : log1pf(__expf(b));
      } else if (EPI == EPI_RES) {
        outF[o] = v + res[o];
      } else if (EPI == EPI_RELU) {
        float b = v + bias[col];
        outH[o] = (half_t)(b > 0.f ? b : 0.f);
      } else { // EPI_BIASRES
        outF[o] = v + bias[col] + res[o];
      }
    }
  }
}

// -------------------- causal depthwise conv (width 4) + SiLU ---------------
__global__ void k_conv_silu(const float* __restrict__ xz, const float* __restrict__ cw,
                            const float* __restrict__ cb,
                            float* __restrict__ uF, half_t* __restrict__ uH) {
  const int idx = blockIdx.x * blockDim.x + threadIdx.x;  // t*D_INNER + d
  const int d = idx & (D_INNER - 1);
  const int t = idx >> 11;
  const int l = t & (SEQL - 1);
  float s = cb[d];
  #pragma unroll
  for (int j = 0; j < 4; ++j) {
    int ll = l - 3 + j;
    if (ll >= 0) s += cw[d * 4 + j] * xz[(size_t)(t - 3 + j) * (2 * D_INNER) + d];
  }
  float v = s / (1.f + __expf(-s));  // silu
  uF[idx] = v;
  uH[idx] = (half_t)v;
}

// -------------------------- selective scan (SSM) ---------------------------
// one lane per (b, d, n) state; 16-lane shfl_xor reduction for y = sum_n s*C.
__global__ void k_ssm_scan(const float* __restrict__ dtF, const float* __restrict__ uF,
                           const float* __restrict__ dbc, const float* __restrict__ xz,
                           const float* __restrict__ A_log, const float* __restrict__ Dsk,
                           half_t* __restrict__ yH) {
  const int gid = blockIdx.x * blockDim.x + threadIdx.x;  // NB*D_INNER*16
  const int n  = gid & 15;
  const int ch = gid >> 4;            // b*D_INNER + d
  const int d  = ch & (D_INNER - 1);
  const int b  = ch >> 11;
  const float Acoef = -__expf(A_log[d * NSTATE + n]);
  const float dval  = Dsk[d];
  float s = 0.f;
  const size_t tbase = (size_t)b * SEQL;
  for (int l = 0; l < SEQL; ++l) {
    const size_t t = tbase + l;
    const float dtv = dtF[t * D_INNER + d];
    const float uv  = uF [t * D_INNER + d];
    const float Bv  = dbc[t * 96 + DTRANK + n];
    const float Cv  = dbc[t * 96 + DTRANK + NSTATE + n];
    s = s * __expf(dtv * Acoef) + dtv * uv * Bv;
    float p = s * Cv;
    p += __shfl_xor(p, 1, 32);
    p += __shfl_xor(p, 2, 32);
    p += __shfl_xor(p, 4, 32);
    p += __shfl_xor(p, 8, 32);
    if (n == 0) {
      const float z = xz[t * (2 * D_INNER) + D_INNER + d];
      const float y = p + uv * dval;
      const float g = z / (1.f + __expf(-z));
      yH[t * D_INNER + d] = (half_t)(y * g);
    }
  }
}

// ------------------------------- host side ---------------------------------
extern "C" void kernel_launch(void* const* d_in, const int* in_sizes, int n_in,
                              void* d_out, int out_size, void* d_ws, size_t ws_size,
                              hipStream_t stream) {
  const float* x        = (const float*)d_in[0];
  const float* rms1_w   = (const float*)d_in[1];
  const float* rms2_w   = (const float*)d_in[2];
  const float* in_proj  = (const float*)d_in[3];
  const float* conv_w   = (const float*)d_in[4];
  const float* conv_b   = (const float*)d_in[5];
  const float* x_proj   = (const float*)d_in[6];
  const float* dt_projw = (const float*)d_in[7];
  const float* dt_b     = (const float*)d_in[8];
  const float* A_log    = (const float*)d_in[9];
  const float* D_skip   = (const float*)d_in[10];
  const float* out_proj = (const float*)d_in[11];
  const float* fc1_w    = (const float*)d_in[12];
  const float* fc1_b    = (const float*)d_in[13];
  const float* fc2_w    = (const float*)d_in[14];
  const float* fc2_b    = (const float*)d_in[15];
  const float* fc3_w    = (const float*)d_in[16];
  const float* fc3_b    = (const float*)d_in[17];
  float* out = (float*)d_out;

  char* ws = (char*)d_ws;
  size_t off = 0;
  auto alloc = [&](size_t bytes) -> void* {
    void* p = ws + off;
    off += (bytes + 255) & ~(size_t)255;
    return p;
  };

  // f16 weights
  half_t* wh_in = (half_t*)alloc((size_t)4096 * 1024 * sizeof(half_t));
  half_t* wh_xp = (half_t*)alloc((size_t)96   * 2048 * sizeof(half_t));
  half_t* wh_dt = (half_t*)alloc((size_t)2048 * 64   * sizeof(half_t));
  half_t* wh_op = (half_t*)alloc((size_t)1024 * 2048 * sizeof(half_t));
  half_t* wh_f1 = (half_t*)alloc((size_t)2048 * 1024 * sizeof(half_t));
  half_t* wh_f2 = (half_t*)alloc((size_t)2048 * 2048 * sizeof(half_t));
  half_t* wh_f3 = (half_t*)alloc((size_t)1024 * 2048 * sizeof(half_t));
  // activations
  half_t* x0h   = (half_t*)alloc((size_t)NTOK * D_MODEL * sizeof(half_t));
  float*  xz    = (float*) alloc((size_t)NTOK * 2 * D_INNER * sizeof(float));
  float*  uF    = (float*) alloc((size_t)NTOK * D_INNER * sizeof(float));
  half_t* uH    = (half_t*)alloc((size_t)NTOK * D_INNER * sizeof(half_t));
  float*  dbc   = (float*) alloc((size_t)NTOK * 96 * sizeof(float));
  half_t* dt64h = (half_t*)alloc((size_t)NTOK * DTRANK * sizeof(half_t));
  float*  dtF   = (float*) alloc((size_t)NTOK * D_INNER * sizeof(float));
  half_t* yH    = (half_t*)alloc((size_t)NTOK * D_INNER * sizeof(half_t));
  float*  xres  = (float*) alloc((size_t)NTOK * D_MODEL * sizeof(float));
  half_t* x2h   = (half_t*)alloc((size_t)NTOK * D_MODEL * sizeof(half_t));
  half_t* h1h   = (half_t*)alloc((size_t)NTOK * 2 * D_MODEL * sizeof(half_t));
  half_t* h2h   = (half_t*)alloc((size_t)NTOK * 2 * D_MODEL * sizeof(half_t));

  auto cvt = [&](const float* src, half_t* dst, int n) {
    k_f32_to_f16<<<(n + 255) / 256, 256, 0, stream>>>(src, dst, n);
  };
  cvt(in_proj,  wh_in, 4096 * 1024);
  cvt(x_proj,   wh_xp, 96 * 2048);
  cvt(dt_projw, wh_dt, 2048 * 64);
  cvt(out_proj, wh_op, 1024 * 2048);
  cvt(fc1_w,    wh_f1, 2048 * 1024);
  cvt(fc2_w,    wh_f2, 2048 * 2048);
  cvt(fc3_w,    wh_f3, 1024 * 2048);

  auto lgrid = [](int M, int N) { return (M >> 7) * (N >> 7); };   // LDS-tiled
  auto dgrid = [](int M, int N) { return ((M >> 6) * (N >> 4) + 7) / 8; };  // direct

  // x0 = rmsnorm(x)
  k_rmsnorm<<<NTOK, 256, 0, stream>>>(x, rms1_w, x0h, D_MODEL);
  // xz = x0 @ in_proj^T   (8192 x 4096 x 1024)
  k_gemm_lds<EPI_F32><<<lgrid(NTOK, 4096), 256, 0, stream>>>(
      x0h, wh_in, xz, nullptr, nullptr, nullptr, NTOK, 4096, 1024);
  // u = silu(causal_conv(u_raw))
  k_conv_silu<<<(NTOK * D_INNER) / 256, 256, 0, stream>>>(xz, conv_w, conv_b, uF, uH);
  // dbc = u @ x_proj^T    (8192 x 96 x 2048) -- N=96, direct kernel
  k_gemm_wmma<EPI_DBC><<<dgrid(NTOK, 96), 256, 0, stream>>>(
      uH, wh_xp, dbc, dt64h, nullptr, nullptr, NTOK, 96, 2048);
  // dt = softplus(dt64 @ dt_proj^T + dt_b)   (8192 x 2048 x 64) -- K=64, direct
  k_gemm_wmma<EPI_SOFTPLUS><<<dgrid(NTOK, 2048), 256, 0, stream>>>(
      dt64h, wh_dt, dtF, nullptr, dt_b, nullptr, NTOK, 2048, 64);
  // selective scan + D-skip + silu(z) gate -> y (f16)
  k_ssm_scan<<<(NB * D_INNER * 16) / 256, 256, 0, stream>>>(
      dtF, uF, dbc, xz, A_log, D_skip, yH);
  // xres = y @ out_proj^T + x   (8192 x 1024 x 2048)
  k_gemm_lds<EPI_RES><<<lgrid(NTOK, 1024), 256, 0, stream>>>(
      yH, wh_op, xres, nullptr, nullptr, x, NTOK, 1024, 2048);
  // x2 = rmsnorm(xres)
  k_rmsnorm<<<NTOK, 256, 0, stream>>>(xres, rms2_w, x2h, D_MODEL);
  // h1 = relu(x2 @ fc1^T + b1)
  k_gemm_lds<EPI_RELU><<<lgrid(NTOK, 2048), 256, 0, stream>>>(
      x2h, wh_f1, nullptr, h1h, fc1_b, nullptr, NTOK, 2048, 1024);
  // h2 = relu(h1 @ fc2^T + b2)
  k_gemm_lds<EPI_RELU><<<lgrid(NTOK, 2048), 256, 0, stream>>>(
      h1h, wh_f2, nullptr, h2h, fc2_b, nullptr, NTOK, 2048, 2048);
  // out = h2 @ fc3^T + b3 + xres
  k_gemm_lds<EPI_BIASRES><<<lgrid(NTOK, 1024), 256, 0, stream>>>(
      h2h, wh_f3, out, nullptr, fc3_b, xres, NTOK, 1024, 2048);
}
